// LinearDecoder_55929064128629
// MI455X (gfx1250) — matrix operand order repaired and checked by reading, not compiled, not run on hardware
//
#include <hip/hip_runtime.h>
#include <hip/hip_bf16.h>
#include <cstddef>

// ---------------------------------------------------------------------------
// LinearDecoder on MI455X (gfx1250): f16 WMMA GEMMs + fused LSTM pipeline.
// All matrix math runs through v_wmma_f32_16x16x32_f16 (wave32).
// GEMM: block tile 256x128, 8 waves (4x2), wave tile 64x64.
// Per K-step: one clause of 16 b128 loads (4 A frags + 4 B frags in distinct
// registers), then 16 back-to-back wmma with staggered s_wait_loadcnt.
// ---------------------------------------------------------------------------

typedef __attribute__((ext_vector_type(16))) _Float16 v16h;
typedef __attribute__((ext_vector_type(8)))  _Float16 v8h;
typedef __attribute__((ext_vector_type(8)))  float    v8f;

#define BATCH  8192
#define LATD   256
#define HIDD   512
#define INPD   128
#define TSTEPS 64
#define MLPW   512
#define GDIM   2048   // 4*HID
#define KCAT   640    // INP + HID

__device__ __forceinline__ float sigf(float x) { return 1.0f / (1.0f + expf(-x)); }

// --------------------------- prep kernels ----------------------------------

// Wt[n*K + k] = (f16) W[k*N + n]   (transpose + downconvert)
__global__ __launch_bounds__(256) void k_convT(const float* __restrict__ W,
                                               _Float16* __restrict__ Wt,
                                               int K, int N) {
  size_t i = (size_t)blockIdx.x * 256 + threadIdx.x;
  if (i >= (size_t)K * N) return;
  int k = (int)(i / N), n = (int)(i % N);
  Wt[(size_t)n * K + k] = (_Float16)W[i];
}

// Wgt[n*640 + k] = k<128 ? Wih[k][n] : Whh[k-128][n]   (fused recurrent weight)
__global__ __launch_bounds__(256) void k_build_wg(const float* __restrict__ Wih,
                                                  const float* __restrict__ Whh,
                                                  _Float16* __restrict__ Wgt) {
  size_t i = (size_t)blockIdx.x * 256 + threadIdx.x;
  if (i >= (size_t)GDIM * KCAT) return;
  int n = (int)(i / KCAT), k = (int)(i % KCAT);
  float v = (k < INPD) ? Wih[(size_t)k * GDIM + n] : Whh[(size_t)(k - INPD) * GDIM + n];
  Wgt[i] = (_Float16)v;
}

__global__ __launch_bounds__(256) void k_bias_sum(const float* __restrict__ a,
                                                  const float* __restrict__ b,
                                                  float* __restrict__ o, int n) {
  int i = blockIdx.x * 256 + threadIdx.x;
  if (i < n) o[i] = a[i] + b[i];
}

__global__ __launch_bounds__(256) void k_f32tof16(const float* __restrict__ a,
                                                  _Float16* __restrict__ o, size_t n) {
  size_t i = (size_t)blockIdx.x * 256 + threadIdx.x;
  if (i < n) o[i] = (_Float16)a[i];
}

// --------------------------- WMMA GEMM -------------------------------------
// C[M x N] = act( A[M x K](f16, row-major, lda) @ Bt[N x K](f16)^T + bias )
// Block tile 256x128, 8 waves of 32 lanes (wm 0..3 x wn 0..1),
// wave tile 64x64: 4x4 16x16 accumulator fragments.
// A frag: lane holds row (lane&15), contiguous K run of 16 at (lane>>4)*16.
// B frag: lane holds col (lane&15), contiguous K run of 16 at (lane>>4)*16.
__global__ __launch_bounds__(256) void k_gemm_wmma(
    const _Float16* __restrict__ A, int lda,
    const _Float16* __restrict__ Bt, int K,
    const float* __restrict__ bias,
    float* __restrict__ C32, int ldc32,
    _Float16* __restrict__ C16, int ldc16,
    int act) {
  const int lane = threadIdx.x & 31;
  const int wave = threadIdx.x >> 5;
  const int wm = wave >> 1;          // 0..3
  const int wn = wave & 1;           // 0..1
  const int m0 = blockIdx.y * 256 + wm * 64;
  const int n0 = blockIdx.x * 128 + wn * 64;
  const int lr = lane & 15;
  const int lh = lane >> 4;          // 0 or 1

  v8f acc[4][4] = {};

  const _Float16* Arow[4];
#pragma unroll
  for (int i = 0; i < 4; i++)
    Arow[i] = A + (size_t)(m0 + i * 16 + lr) * lda + lh * 16;
  const _Float16* Brow[4];
#pragma unroll
  for (int j = 0; j < 4; j++)
    Brow[j] = Bt + (size_t)(n0 + j * 16 + lr) * K + lh * 16;

  for (int k0 = 0; k0 < K; k0 += 32) {
    // Gather ALL operand fragments for this K-step first, in distinct
    // registers, so the scheduler can emit one load clause and cover the
    // latency with staggered partial loadcnt waits.
    v16h afrag[4];
    v16h bfrag[4];
#pragma unroll
    for (int i = 0; i < 4; i++) {
      v8h alo = *(const v8h*)(Arow[i] + k0);
      v8h ahi = *(const v8h*)(Arow[i] + k0 + 8);
#pragma unroll
      for (int e = 0; e < 8; e++) { afrag[i][e] = alo[e]; afrag[i][8 + e] = ahi[e]; }
    }
#pragma unroll
    for (int j = 0; j < 4; j++)
      bfrag[j] = *(const v16h*)(Brow[j] + k0);

#pragma unroll
    for (int j = 0; j < 4; j++) {
#pragma unroll
      for (int i = 0; i < 4; i++) {
        acc[i][j] = __builtin_amdgcn_wmma_f32_16x16x32_f16(
            false, afrag[i], false, bfrag[j], (short)0, acc[i][j], false, false);
      }
    }
  }

#pragma unroll
  for (int j = 0; j < 4; j++) {
    int col = n0 + j * 16 + lr;
    float bb = bias ? bias[col] : 0.0f;
#pragma unroll
    for (int i = 0; i < 4; i++) {
#pragma unroll
      for (int r = 0; r < 8; r++) {
        int row = m0 + i * 16 + lh * 8 + r;
        float v = acc[i][j][r] + bb;
        if (act) v = (v >= 0.0f) ? v : 0.2f * v;       // leaky relu 0.2
        if (C32) C32[(size_t)row * ldc32 + col] = v;
        if (C16) C16[(size_t)row * ldc16 + col] = (_Float16)v;
      }
    }
  }
}

// --------------------------- LSTM cell -------------------------------------
// gates [B x 2048] (i,f,g,o); c updated in place; h2 written f16 into xcat.
__global__ __launch_bounds__(256) void k_lstm_cell(const float* __restrict__ gates,
                                                   float* __restrict__ c,
                                                   _Float16* __restrict__ xcat) {
  size_t idx = (size_t)blockIdx.x * 256 + threadIdx.x;   // over B*HID
  if (idx >= (size_t)BATCH * HIDD) return;
  int b = (int)(idx >> 9), j = (int)(idx & (HIDD - 1));
  const float* g = gates + (size_t)b * GDIM;
  float gi = sigf(g[j]);
  float gf = sigf(g[HIDD + j]);
  float gg = tanhf(g[2 * HIDD + j]);
  float go = sigf(g[3 * HIDD + j]);
  float c2 = gf * c[idx] + gi * gg;
  float h2 = go * tanhf(c2);
  c[idx] = c2;
  xcat[(size_t)b * KCAT + INPD + j] = (_Float16)h2;
}

// --------------------------- head: softmax(127)+sigmoid(1) -----------------
// One wave (32 lanes, 4 values/lane) per row of 128 logits.
__global__ __launch_bounds__(256) void k_head(const float* __restrict__ logits,
                                              float* __restrict__ out,     // may be null
                                              size_t out_row_stride,
                                              _Float16* __restrict__ xcat) {
  int row = blockIdx.x * 8 + (threadIdx.x >> 5);
  int lane = threadIdx.x & 31;
  const float* rp = logits + (size_t)row * INPD;
  float v[4];
#pragma unroll
  for (int e = 0; e < 4; e++) v[e] = rp[lane * 4 + e];

  float m = -1e30f;
#pragma unroll
  for (int e = 0; e < 4; e++) if (lane * 4 + e < INPD - 1) m = fmaxf(m, v[e]);
#pragma unroll
  for (int o = 16; o > 0; o >>= 1) m = fmaxf(m, __shfl_xor(m, o, 32));

  float s = 0.0f;
#pragma unroll
  for (int e = 0; e < 4; e++) if (lane * 4 + e < INPD - 1) s += expf(v[e] - m);
#pragma unroll
  for (int o = 16; o > 0; o >>= 1) s += __shfl_xor(s, o, 32);
  float inv = 1.0f / s;

#pragma unroll
  for (int e = 0; e < 4; e++) {
    int idx = lane * 4 + e;
    float p = (idx < INPD - 1) ? expf(v[e] - m) * inv : sigf(v[e]);
    if (out) out[(size_t)row * out_row_stride + idx] = p;
    xcat[(size_t)row * KCAT + idx] = (_Float16)p;
  }
}

// --------------------------- launcher --------------------------------------

extern "C" void kernel_launch(void* const* d_in, const int* in_sizes, int n_in,
                              void* d_out, int out_size, void* d_ws, size_t ws_size,
                              hipStream_t stream) {
  (void)in_sizes; (void)n_in; (void)out_size; (void)ws_size;

  const float* x   = (const float*)d_in[0];
  const float* W1  = (const float*)d_in[1];  const float* b1  = (const float*)d_in[2];
  const float* W2  = (const float*)d_in[3];  const float* b2  = (const float*)d_in[4];
  const float* W3  = (const float*)d_in[5];  const float* b3  = (const float*)d_in[6];
  const float* Wh1 = (const float*)d_in[7];  const float* bh1 = (const float*)d_in[8];
  const float* Wh2 = (const float*)d_in[9];  const float* bh2 = (const float*)d_in[10];
  const float* Wc1 = (const float*)d_in[11]; const float* bc1 = (const float*)d_in[12];
  const float* Wc2 = (const float*)d_in[13]; const float* bc2 = (const float*)d_in[14];
  const float* Wx1 = (const float*)d_in[15]; const float* bx1 = (const float*)d_in[16];
  const float* Wx2 = (const float*)d_in[17]; const float* bx2 = (const float*)d_in[18];
  const float* Wih = (const float*)d_in[19]; const float* bih = (const float*)d_in[20];
  const float* Whh = (const float*)d_in[21]; const float* bhh = (const float*)d_in[22];
  const float* Wp  = (const float*)d_in[23]; const float* bp  = (const float*)d_in[24];
  float* out = (float*)d_out;

  // ---- workspace carve-up (256B aligned) ----
  char* w = (char*)d_ws;
  auto alloc = [&](size_t bytes) -> void* {
    void* p = (void*)w;
    w += (bytes + 255) & ~(size_t)255;
    return p;
  };
  _Float16* W1t  = (_Float16*)alloc((size_t)LATD * MLPW * 2);
  _Float16* W2t  = (_Float16*)alloc((size_t)MLPW * MLPW * 2);
  _Float16* W3t  = (_Float16*)alloc((size_t)MLPW * MLPW * 2);
  _Float16* Wh1t = (_Float16*)alloc((size_t)MLPW * MLPW * 2);
  _Float16* Wh2t = (_Float16*)alloc((size_t)MLPW * HIDD * 2);
  _Float16* Wc1t = (_Float16*)alloc((size_t)MLPW * MLPW * 2);
  _Float16* Wc2t = (_Float16*)alloc((size_t)MLPW * HIDD * 2);
  _Float16* Wx1t = (_Float16*)alloc((size_t)MLPW * MLPW * 2);
  _Float16* Wx2t = (_Float16*)alloc((size_t)MLPW * INPD * 2);
  _Float16* Wpt  = (_Float16*)alloc((size_t)HIDD * INPD * 2);
  _Float16* Wgt  = (_Float16*)alloc((size_t)GDIM * KCAT * 2);
  float*    bg   = (float*)   alloc((size_t)GDIM * 4);
  _Float16* xf16 = (_Float16*)alloc((size_t)BATCH * LATD * 2);
  _Float16* zA   = (_Float16*)alloc((size_t)BATCH * MLPW * 2);
  _Float16* zB   = (_Float16*)alloc((size_t)BATCH * MLPW * 2);
  _Float16* xcat = (_Float16*)alloc((size_t)BATCH * KCAT * 2);   // [xt | h] f16
  float*    cbuf = (float*)   alloc((size_t)BATCH * HIDD * 4);
  float*    gate = (float*)   alloc((size_t)BATCH * GDIM * 4);
  float*    lgt  = (float*)   alloc((size_t)BATCH * INPD * 4);

  auto gblk = [](size_t n) { return dim3((unsigned)((n + 255) / 256)); };

  // ---- weight prep (f16, transposed to [N x K]) ----
  k_convT<<<gblk((size_t)LATD * MLPW), 256, 0, stream>>>(W1,  W1t,  LATD, MLPW);
  k_convT<<<gblk((size_t)MLPW * MLPW), 256, 0, stream>>>(W2,  W2t,  MLPW, MLPW);
  k_convT<<<gblk((size_t)MLPW * MLPW), 256, 0, stream>>>(W3,  W3t,  MLPW, MLPW);
  k_convT<<<gblk((size_t)MLPW * MLPW), 256, 0, stream>>>(Wh1, Wh1t, MLPW, MLPW);
  k_convT<<<gblk((size_t)MLPW * HIDD), 256, 0, stream>>>(Wh2, Wh2t, MLPW, HIDD);
  k_convT<<<gblk((size_t)MLPW * MLPW), 256, 0, stream>>>(Wc1, Wc1t, MLPW, MLPW);
  k_convT<<<gblk((size_t)MLPW * HIDD), 256, 0, stream>>>(Wc2, Wc2t, MLPW, HIDD);
  k_convT<<<gblk((size_t)MLPW * MLPW), 256, 0, stream>>>(Wx1, Wx1t, MLPW, MLPW);
  k_convT<<<gblk((size_t)MLPW * INPD), 256, 0, stream>>>(Wx2, Wx2t, MLPW, INPD);
  k_convT<<<gblk((size_t)HIDD * INPD), 256, 0, stream>>>(Wp,  Wpt,  HIDD, INPD);
  k_build_wg<<<gblk((size_t)GDIM * KCAT), 256, 0, stream>>>(Wih, Whh, Wgt);
  k_bias_sum<<<gblk(GDIM), 256, 0, stream>>>(bih, bhh, bg, GDIM);
  k_f32tof16<<<gblk((size_t)BATCH * LATD), 256, 0, stream>>>(x, xf16, (size_t)BATCH * LATD);

  const dim3 blk(256);
  auto ggrid = [](int N) { return dim3(N / 128, BATCH / 256); };

  // ---- MLP trunk ----
  k_gemm_wmma<<<ggrid(MLPW), blk, 0, stream>>>(xf16, LATD, W1t, LATD, b1,
                                               nullptr, 0, zA, MLPW, 1);
  k_gemm_wmma<<<ggrid(MLPW), blk, 0, stream>>>(zA, MLPW, W2t, MLPW, b2,
                                               nullptr, 0, zB, MLPW, 1);
  k_gemm_wmma<<<ggrid(MLPW), blk, 0, stream>>>(zB, MLPW, W3t, MLPW, b3,
                                               nullptr, 0, zA, MLPW, 1);   // z3 -> zA
  // h branch -> f16 straight into xcat[:, 128:640]
  k_gemm_wmma<<<ggrid(MLPW), blk, 0, stream>>>(zA, MLPW, Wh1t, MLPW, bh1,
                                               nullptr, 0, zB, MLPW, 1);
  k_gemm_wmma<<<ggrid(HIDD), blk, 0, stream>>>(zB, MLPW, Wh2t, MLPW, bh2,
                                               nullptr, 0, xcat + INPD, KCAT, 0);
  // c branch -> f32 state
  k_gemm_wmma<<<ggrid(MLPW), blk, 0, stream>>>(zA, MLPW, Wc1t, MLPW, bc1,
                                               nullptr, 0, zB, MLPW, 1);
  k_gemm_wmma<<<ggrid(HIDD), blk, 0, stream>>>(zB, MLPW, Wc2t, MLPW, bc2,
                                               cbuf, HIDD, nullptr, 0, 0);
  // x0 branch -> head -> xcat[:, 0:128]
  k_gemm_wmma<<<ggrid(MLPW), blk, 0, stream>>>(zA, MLPW, Wx1t, MLPW, bx1,
                                               nullptr, 0, zB, MLPW, 1);
  k_gemm_wmma<<<ggrid(INPD), blk, 0, stream>>>(zB, MLPW, Wx2t, MLPW, bx2,
                                               lgt, INPD, nullptr, 0, 0);
  k_head<<<dim3(BATCH / 8), blk, 0, stream>>>(lgt, nullptr, 0, xcat);

  // ---- autoregressive LSTM, 64 steps ----
  for (int t = 0; t < TSTEPS; t++) {
    // gates = [xt|h] @ [Wih;Whh] + (bih+bhh)
    k_gemm_wmma<<<ggrid(GDIM), blk, 0, stream>>>(xcat, KCAT, Wgt, KCAT, bg,
                                                 gate, GDIM, nullptr, 0, 0);
    k_lstm_cell<<<gblk((size_t)BATCH * HIDD), 256, 0, stream>>>(gate, cbuf, xcat);
    // logits = h2 @ Wp + bp   (h2 read strided out of xcat)
    k_gemm_wmma<<<ggrid(INPD), blk, 0, stream>>>(xcat + INPD, KCAT, Wpt, HIDD, bp,
                                                 lgt, INPD, nullptr, 0, 0);
    // head -> ys[:, t, :] (f32) and next xt (f16)
    k_head<<<dim3(BATCH / 8), blk, 0, stream>>>(lgt, out + (size_t)t * INPD,
                                                (size_t)TSTEPS * INPD, xcat);
  }
}
